// StructureModule_43439299232377
// MI455X (gfx1250) — compile-verified, matrix operand order checked by reading, not run on hardware
//
#include <hip/hip_runtime.h>
#include <math.h>

typedef __attribute__((ext_vector_type(16))) _Float16 v16h;
typedef __attribute__((ext_vector_type(8)))  _Float16 v8h;
typedef __attribute__((ext_vector_type(8)))  float    v8f;

#define NRES 512
#define CS_ 384
#define CZ_ 128
#define CI_ 16
#define H_ 12
#define PQ_ 4
#define PV_ 8
#define CRES_ 128
#define NANG_ 7
#define CAT_ 2112

#define WC_      0.235702260395516f   /* sqrt(2/(9*PQ)) */
#define INV_S3   0.577350269189626f
#define RSQRT48  0.144337567297406f

// ---------------- weight convert+transpose: wt[n][k] = (f16) w[k][n]  (one-time)
__global__ void wconv_kernel(const float* __restrict__ w, _Float16* __restrict__ wt, int K, int N)
{
    int idx = blockIdx.x * 256 + threadIdx.x;
    if (idx >= K * N) return;
    int n = idx / K, k = idx - n * K;
    wt[(size_t)n * K + k] = (_Float16)w[(size_t)k * N + n];
}

// ---------------- WMMA GEMM: out[M,N] = (relu?)A[M,K] @ W[K,N] + bias (+add) (relu?)
// Wt is f16, transposed [N][K]. 4 waves/block; each wave: 16 rows x NT 16-col tiles.
// A fragment: 4x b128 f32 loads; B fragment: 2x b128 f16 loads. K % 32 == 0.
template <int NT, int FLAGS, bool HASADD>
__global__ void wmma_gemm_t(const float* __restrict__ A, const _Float16* __restrict__ Wt,
                            const float* __restrict__ bias, const float* __restrict__ addsrc,
                            float* __restrict__ out, int M, int K, int Nc)
{
    int lane = threadIdx.x & 31;
    int wave = threadIdx.x >> 5;
    int row0 = (blockIdx.y * 4 + wave) * 16;
    int col0 = blockIdx.x * (16 * NT);
    int mrow = lane & 15;          // A row / B col owned by this lane
    int kb   = (lane >> 4) * 8;    // K sub-block select
    v8f acc[NT];
#pragma unroll
    for (int t = 0; t < NT; ++t) acc[t] = (v8f){};
    const float* Arow = A + (size_t)(row0 + mrow) * K;
    const _Float16* Wcol[NT];
#pragma unroll
    for (int t = 0; t < NT; ++t) Wcol[t] = Wt + (size_t)(col0 + 16 * t + mrow) * K;

#pragma unroll 2
    for (int k0 = 0; k0 < K; k0 += 32) {
        const float4* ap0 = reinterpret_cast<const float4*>(Arow + k0 + kb);
        const float4* ap1 = reinterpret_cast<const float4*>(Arow + k0 + 16 + kb);
        float a16[16];
        *(float4*)(a16 + 0)  = ap0[0];
        *(float4*)(a16 + 4)  = ap0[1];
        *(float4*)(a16 + 8)  = ap1[0];
        *(float4*)(a16 + 12) = ap1[1];
        v16h af;
#pragma unroll
        for (int e = 0; e < 16; ++e) {
            float v = a16[e];
            if (FLAGS & 1) v = fmaxf(v, 0.f);
            af[e] = (_Float16)v;
        }
#pragma unroll
        for (int t = 0; t < NT; ++t) {
            v8h blo = *reinterpret_cast<const v8h*>(Wcol[t] + k0 + kb);
            v8h bhi = *reinterpret_cast<const v8h*>(Wcol[t] + k0 + 16 + kb);
            v16h bf = __builtin_shufflevector(blo, bhi,
                        0,1,2,3,4,5,6,7,8,9,10,11,12,13,14,15);
            acc[t] = __builtin_amdgcn_wmma_f32_16x16x32_f16(false, af, false, bf,
                                                            (short)0, acc[t], false, false);
        }
    }
    int ncol  = lane & 15;
    int mbase = (lane >> 4) * 8;
#pragma unroll
    for (int t = 0; t < NT; ++t) {
#pragma unroll
        for (int v = 0; v < 8; ++v) {
            int r = row0 + v + mbase;
            int c = col0 + 16 * t + ncol;
            float val = acc[t][v] + bias[c];
            if (HASADD) val += addsrc[(size_t)r * Nc + c];
            if (FLAGS & 2) val = fmaxf(val, 0.f);
            out[(size_t)r * Nc + c] = val;
        }
    }
}

// ---------------- LayerNorm (one block per row)
__global__ void ln_kernel(const float* __restrict__ x, const float* __restrict__ g,
                          const float* __restrict__ b, float* __restrict__ y, int cols)
{
    int row = blockIdx.x, tid = threadIdx.x;
    __shared__ float rs[8], rs2[8], mv[2];
    const float* xr = x + (size_t)row * cols;
    float s = 0.f, s2 = 0.f;
    for (int c = tid; c < cols; c += blockDim.x) { float v = xr[c]; s += v; s2 += v*v; }
    for (int off = 16; off; off >>= 1) { s += __shfl_xor(s, off); s2 += __shfl_xor(s2, off); }
    if ((tid & 31) == 0) { rs[tid>>5] = s; rs2[tid>>5] = s2; }
    __syncthreads();
    if (tid == 0) {
        float S = 0.f, S2 = 0.f; int nw = blockDim.x >> 5;
        for (int k = 0; k < nw; ++k) { S += rs[k]; S2 += rs2[k]; }
        float m = S / cols, var = S2 / cols - m * m;
        mv[0] = m; mv[1] = rsqrtf(var + 1e-5f);
    }
    __syncthreads();
    float m = mv[0], r = mv[1];
    float* yr = y + (size_t)row * cols;
    for (int c = tid; c < cols; c += blockDim.x) yr[c] = (xr[c] - m) * r * g[c] + b[c];
}

// ---------------- per-(i,j) LN stats of z (128 channels)
__global__ void zstats_kernel(const float* __restrict__ z, float* __restrict__ zm,
                              float* __restrict__ zr)
{
    size_t p = blockIdx.x; int tid = threadIdx.x; // 128 threads
    float v = z[p * 128 + tid];
    float s = v, s2 = v * v;
    for (int off = 16; off; off >>= 1) { s += __shfl_xor(s, off); s2 += __shfl_xor(s2, off); }
    __shared__ float a[4], b2[4];
    if ((tid & 31) == 0) { a[tid>>5] = s; b2[tid>>5] = s2; }
    __syncthreads();
    if (tid == 0) {
        float S = a[0]+a[1]+a[2]+a[3], S2 = b2[0]+b2[1]+b2[2]+b2[3];
        float m = S / 128.f, var = S2 / 128.f - m * m;
        zm[p] = m; zr[p] = rsqrtf(var + 1e-5f);
    }
}

// ---------------- fold LN(z) into bias projection
__global__ void biasprep_kernel(const float* __restrict__ bias_w, const float* __restrict__ bias_b,
                                const float* __restrict__ lnzg, const float* __restrict__ lnzb,
                                float* __restrict__ gW, float* __restrict__ colGW, float* __restrict__ bW)
{
    int tid = threadIdx.x;
    for (int idx = tid; idx < 128 * 12; idx += 256) {
        int ch = idx / 12, hh = idx % 12;
        gW[idx] = lnzg[ch] * bias_w[ch * 12 + hh];
    }
    __syncthreads();
    if (tid < 12) {
        float cs = 0.f, bs = 0.f;
        for (int ch = 0; ch < 128; ++ch) { cs += gW[ch*12+tid]; bs += lnzb[ch] * bias_w[ch*12+tid]; }
        colGW[tid] = cs; bW[tid] = bs + bias_b[tid];
    }
}

// ---------------- biasB[h,i,j] = LN(z[i,j]) @ bias_w + bias_b  (one wave per (i,j))
__global__ void bias_kernel(const float* __restrict__ z, const float* __restrict__ zm,
                            const float* __restrict__ zr, const float* __restrict__ gW,
                            const float* __restrict__ colGW, const float* __restrict__ bW,
                            float* __restrict__ biasB)
{
    __shared__ float gWs[128 * 12];
    __shared__ float cg[12], bw[12];
    int tid = threadIdx.x;
    for (int idx = tid; idx < 1536; idx += 256) gWs[idx] = gW[idx];
    if (tid < 12) { cg[tid] = colGW[tid]; bw[tid] = bW[tid]; }
    __syncthreads();
    int wave = tid >> 5, lane = tid & 31;
    size_t pair = (size_t)blockIdx.x * 8 + wave;
    int i = (int)(pair >> 9), j = (int)(pair & 511);
    const float* zrow = &z[pair * 128];
    float part[12];
#pragma unroll
    for (int h = 0; h < 12; ++h) part[h] = 0.f;
    for (int cc = 0; cc < 4; ++cc) {
        int ch = lane * 4 + cc;
        float zv = zrow[ch];
#pragma unroll
        for (int h = 0; h < 12; ++h) part[h] += zv * gWs[ch * 12 + h];
    }
#pragma unroll
    for (int h = 0; h < 12; ++h)
        for (int off = 16; off; off >>= 1) part[h] += __shfl_xor(part[h], off);
    if (lane == 0) {
        float m = zm[pair], r = zr[pair];
        for (int h = 0; h < 12; ++h)
            biasB[((size_t)h * NRES + i) * NRES + j] = r * part[h] - r * m * cg[h] + bw[h];
    }
}

// ---------------- quat -> rotation matrix
__global__ void rot_kernel(const float* __restrict__ quat, float* __restrict__ Rb)
{
    int i = blockIdx.x * blockDim.x + threadIdx.x;
    if (i >= NRES) return;
    float w = quat[i*4], x = quat[i*4+1], y = quat[i*4+2], z = quat[i*4+3];
    float* R = &Rb[i*9];
    R[0] = 1-2*(y*y+z*z); R[1] = 2*(x*y-w*z);   R[2] = 2*(x*z+w*y);
    R[3] = 2*(x*y+w*z);   R[4] = 1-2*(x*x+z*z); R[5] = 2*(y*z-w*x);
    R[6] = 2*(x*z-w*y);   R[7] = 2*(y*z+w*x);   R[8] = 1-2*(x*x+y*y);
}

__global__ void init_state_kernel(float* __restrict__ quat, float* __restrict__ trans)
{
    int i = blockIdx.x * blockDim.x + threadIdx.x;
    if (i >= NRES) return;
    quat[i*4] = 1.f; quat[i*4+1] = 0.f; quat[i*4+2] = 0.f; quat[i*4+3] = 0.f;
    trans[i*3] = 0.f; trans[i*3+1] = 0.f; trans[i*3+2] = 0.f;
}

// ---------------- rotate projected points to global frame; per-head sq-norm of first 4 points
__global__ void pts_kernel(const float* __restrict__ lin, const float* __restrict__ Rb,
                           const float* __restrict__ trans, float* __restrict__ pts,
                           float* __restrict__ sqBuf, int HP, int nper)
{
    int i = blockIdx.x, tid = threadIdx.x; // blockDim >= HP
    __shared__ float sq[160];
    __shared__ float R[9], tr[3];
    if (tid < 9) R[tid] = Rb[i*9 + tid];
    if (tid < 3) tr[tid] = trans[i*3 + tid];
    __syncthreads();
    if (tid < HP) {
        float x = lin[(size_t)i*3*HP + tid];
        float y = lin[(size_t)i*3*HP + HP + tid];
        float z = lin[(size_t)i*3*HP + 2*HP + tid];
        float gx = R[0]*x + R[1]*y + R[2]*z + tr[0];
        float gy = R[3]*x + R[4]*y + R[5]*z + tr[1];
        float gz = R[6]*x + R[7]*y + R[8]*z + tr[2];
        size_t b = ((size_t)i*HP + tid) * 3;
        pts[b] = gx; pts[b+1] = gy; pts[b+2] = gz;
        int pp = tid % nper;
        sq[tid] = (pp < 4) ? (gx*gx + gy*gy + gz*gz) : 0.f;
    }
    __syncthreads();
    if (tid < 12) {
        float s = 0.f;
        for (int pp = 0; pp < 4; ++pp) s += sq[tid * nper + pp];
        sqBuf[i*12 + tid] = s;
    }
}

// ---------------- attention core: logits + softmax + o + op  (one block per (i,h))
__global__ void attn_kernel(const float* __restrict__ q, const float* __restrict__ kv,
                            const float* __restrict__ qpPts, const float* __restrict__ kvpPts,
                            const float* __restrict__ sqq, const float* __restrict__ sqk,
                            const float* __restrict__ biasB, const float* __restrict__ gamma,
                            const float* __restrict__ mask, float* __restrict__ aBuf,
                            float* __restrict__ oBuf, float* __restrict__ opBuf)
{
    int i = blockIdx.x, h = blockIdx.y, tid = threadIdx.x; // 256 threads
    __shared__ float lg[NRES];
    __shared__ float qs[16], qp_s[12], red[8];
    if (tid < 16) qs[tid] = q[((size_t)i*12 + h)*16 + tid];
    if (tid >= 16 && tid < 28) qp_s[tid-16] = qpPts[((size_t)i*12 + h)*12 + (tid-16)];
    __syncthreads();
    float hw = log1pf(expf(gamma[h])) * WC_ * INV_S3;
    float sqqv = sqq[i*12 + h];
    float maski = mask[i];
    for (int j = tid; j < NRES; j += 256) {
        const float* kj = &kv[((size_t)j*12 + h)*32];
        float qk = 0.f;
#pragma unroll
        for (int c = 0; c < 16; ++c) qk += qs[c] * kj[c];
        const float* kpj = &kvpPts[((size_t)j*12 + h)*36];
        float cross = 0.f;
#pragma unroll
        for (int e = 0; e < 12; ++e) cross += qp_s[e] * kpj[e];
        float d2 = sqqv + sqk[j*12 + h] - 2.f * cross;
        float mb = 100000.f * (maski * mask[j] - 1.f);
        lg[j] = qk * RSQRT48 + INV_S3 * biasB[((size_t)h*NRES + i)*NRES + j]
                - 0.5f * d2 * hw + mb;
    }
    __syncthreads();
    float lmax = -1e30f;
    for (int j = tid; j < NRES; j += 256) lmax = fmaxf(lmax, lg[j]);
    for (int off = 16; off; off >>= 1) lmax = fmaxf(lmax, __shfl_xor(lmax, off));
    if ((tid & 31) == 0) red[tid>>5] = lmax;
    __syncthreads();
    if (tid < 8) { float v = red[tid]; for (int off = 4; off; off >>= 1) v = fmaxf(v, __shfl_xor(v, off)); if (tid == 0) red[0] = v; }
    __syncthreads();
    lmax = red[0];
    __syncthreads();
    float lsum = 0.f;
    for (int j = tid; j < NRES; j += 256) { float e = expf(lg[j] - lmax); lg[j] = e; lsum += e; }
    for (int off = 16; off; off >>= 1) lsum += __shfl_xor(lsum, off);
    if ((tid & 31) == 0) red[tid>>5] = lsum;
    __syncthreads();
    if (tid < 8) { float v = red[tid]; for (int off = 4; off; off >>= 1) v += __shfl_xor(v, off); if (tid == 0) red[0] = v; }
    __syncthreads();
    float inv = 1.f / red[0];
    for (int j = tid; j < NRES; j += 256) {
        float av = lg[j] * inv;
        lg[j] = av;
        aBuf[((size_t)h*NRES + i)*NRES + j] = av;
    }
    __syncthreads();
    int wave = tid >> 5, lane = tid & 31;
    for (int o = wave; o < 40; o += 8) {
        float acc = 0.f;
        if (o < 16) {
            for (int j = lane; j < NRES; j += 32)
                acc += lg[j] * kv[((size_t)j*12 + h)*32 + 16 + o];
        } else {
            int e = o - 16;
            for (int j = lane; j < NRES; j += 32)
                acc += lg[j] * kvpPts[((size_t)j*12 + h)*36 + 12 + e];
        }
        for (int off = 16; off; off >>= 1) acc += __shfl_xor(acc, off);
        if (lane == 0) {
            if (o < 16) oBuf[((size_t)i*12 + h)*16 + o] = acc;
            else        opBuf[((size_t)i*12 + h)*24 + (o - 16)] = acc;
        }
    }
}

// ---------------- o_pair[i,h,c] = sum_j a[h,i,j] * LN(z[i,j])[c], LN folded via (m,r)
__global__ void opair_kernel(const float* __restrict__ z, const float* __restrict__ zm,
                             const float* __restrict__ zr, const float* __restrict__ aBuf,
                             const float* __restrict__ lnzg, const float* __restrict__ lnzb,
                             float* __restrict__ opair)
{
    int i = blockIdx.x, tid = threadIdx.x; // 256
    int c = tid & 127, hg = tid >> 7;
    __shared__ float wsm[12 * 64];
    __shared__ float Tsh[12];
    float acc[6] = {0,0,0,0,0,0};
    for (int j0 = 0; j0 < NRES; j0 += 64) {
        __syncthreads();
        for (int idx = tid; idx < 12 * 64; idx += 256) {
            int h = idx >> 6, jj = idx & 63, j = j0 + jj;
            wsm[idx] = aBuf[((size_t)h*NRES + i)*NRES + j] * zr[(size_t)i*NRES + j];
        }
        __syncthreads();
        const float* zrow = &z[((size_t)i*NRES + j0)*128 + c];
        for (int jj = 0; jj < 64; ++jj) {
            float zv = zrow[(size_t)jj * 128];
            int base = hg * 6 * 64 + jj;
            acc[0] += wsm[base        ] * zv;
            acc[1] += wsm[base + 64   ] * zv;
            acc[2] += wsm[base + 128  ] * zv;
            acc[3] += wsm[base + 192  ] * zv;
            acc[4] += wsm[base + 256  ] * zv;
            acc[5] += wsm[base + 320  ] * zv;
        }
    }
    __syncthreads();
    int wave = tid >> 5, lane = tid & 31;
    for (int h = wave; h < 12; h += 8) {
        float t = 0.f;
        for (int j = lane; j < NRES; j += 32)
            t += aBuf[((size_t)h*NRES + i)*NRES + j] * zr[(size_t)i*NRES + j] * zm[(size_t)i*NRES + j];
        for (int off = 16; off; off >>= 1) t += __shfl_xor(t, off);
        if (lane == 0) Tsh[h] = t;
    }
    __syncthreads();
    float g = lnzg[c], b = lnzb[c];
    for (int k2 = 0; k2 < 6; ++k2) {
        int h = hg * 6 + k2;
        opair[(size_t)i*1536 + h*128 + c] = g * acc[k2] - g * Tsh[h] + b;
    }
}

// ---------------- assemble concat vector: [o | op_x | op_y | op_z | ||op|| | o_pair]
__global__ void cat_kernel(const float* __restrict__ oBuf, const float* __restrict__ opBuf,
                           const float* __restrict__ opair, const float* __restrict__ Rb,
                           const float* __restrict__ trans, float* __restrict__ cat)
{
    int i = blockIdx.x, tid = threadIdx.x; // 256
    __shared__ float R[9], tr[3];
    if (tid < 9) R[tid] = Rb[i*9 + tid];
    if (tid >= 9 && tid < 12) tr[tid-9] = trans[i*3 + tid - 9];
    __syncthreads();
    float* crow = &cat[(size_t)i * CAT_];
    for (int c = tid; c < 192; c += 256) crow[c] = oBuf[(size_t)i*192 + c];
    for (int t = tid; t < 96; t += 256) {
        const float* p = &opBuf[(size_t)i*288 + t*3];
        float vx = p[0]-tr[0], vy = p[1]-tr[1], vz = p[2]-tr[2];
        float lx = R[0]*vx + R[3]*vy + R[6]*vz;   // R^T
        float ly = R[1]*vx + R[4]*vy + R[7]*vz;
        float lz = R[2]*vx + R[5]*vy + R[8]*vz;
        crow[192+t] = lx; crow[288+t] = ly; crow[384+t] = lz;
        crow[480+t] = fmaxf(sqrtf(lx*lx + ly*ly + lz*lz), 1e-6f);
    }
    for (int c = tid; c < 1536; c += 256) crow[576+c] = opair[(size_t)i*1536 + c];
}

// ---------------- backbone update
__global__ void bb_kernel(const float* __restrict__ s, const float* __restrict__ bb_w,
                          const float* __restrict__ bb_b, const float* __restrict__ Rb,
                          float* __restrict__ quat, float* __restrict__ trans)
{
    int i = blockIdx.x, tid = threadIdx.x; // 192 = 6 waves
    int wave = tid >> 5, lane = tid & 31;
    __shared__ float upd[6];
    float acc = 0.f;
    const float* srow = &s[(size_t)i * CS_];
    for (int k = lane; k < CS_; k += 32) acc += srow[k] * bb_w[k*6 + wave];
    for (int off = 16; off; off >>= 1) acc += __shfl_xor(acc, off);
    if (lane == 0) upd[wave] = acc + bb_b[wave];
    __syncthreads();
    if (tid == 0) {
        float w = quat[i*4], x = quat[i*4+1], y = quat[i*4+2], z = quat[i*4+3];
        float b = upd[0], c = upd[1], d = upd[2];
        float mw = -x*b - y*c - z*d;
        float mx =  w*b + y*d - z*c;
        float my =  w*c - x*d + z*b;
        float mz =  w*d + x*c - y*b;
        float nw = w+mw, nx = x+mx, ny = y+my, nz = z+mz;
        float inv = rsqrtf(nw*nw + nx*nx + ny*ny + nz*nz);
        quat[i*4] = nw*inv; quat[i*4+1] = nx*inv; quat[i*4+2] = ny*inv; quat[i*4+3] = nz*inv;
        const float* R = &Rb[i*9];
        float ux = upd[3], uy = upd[4], uz = upd[5];
        trans[i*3+0] += R[0]*ux + R[1]*uy + R[2]*uz;
        trans[i*3+1] += R[3]*ux + R[4]*uy + R[5]*uz;
        trans[i*3+2] += R[6]*ux + R[7]*uy + R[8]*uz;
    }
}

// ---------------- small GEMM w/ relu input (one wave per output column)
__global__ void smallgemm_relu_kernel(const float* __restrict__ A, const float* __restrict__ W,
                                      const float* __restrict__ bias, float* __restrict__ out,
                                      int K, int Nc)
{
    int i = blockIdx.x, tid = threadIdx.x;
    int o = tid >> 5, lane = tid & 31;
    if (o >= Nc) return;
    float acc = 0.f;
    const float* ar = &A[(size_t)i * K];
    for (int k = lane; k < K; k += 32) acc += fmaxf(ar[k], 0.f) * W[k*Nc + o];
    for (int off = 16; off; off >>= 1) acc += __shfl_xor(acc, off);
    if (lane == 0) out[(size_t)i*Nc + o] = acc + bias[o];
}

// ---------------- final output: [quat(4), 10*trans(3), ang(14), ang_n(14), s(384)]
__global__ void write_out_kernel(const float* __restrict__ quat, const float* __restrict__ trans,
                                 const float* __restrict__ ang, const float* __restrict__ s,
                                 float* __restrict__ out, int blk)
{
    int i = blockIdx.x, tid = threadIdx.x; // 256
    float* orow = &out[((size_t)blk * NRES + i) * 419];
    if (tid < 4) orow[tid] = quat[i*4 + tid];
    else if (tid < 7) orow[tid] = 10.f * trans[i*3 + (tid-4)];
    else if (tid < 21) orow[tid] = ang[i*14 + (tid-7)];
    else if (tid < 35) {
        int jj = tid - 21, pr = jj >> 1;
        float a0 = ang[i*14 + 2*pr], a1 = ang[i*14 + 2*pr + 1];
        float n = sqrtf(fmaxf(a0*a0 + a1*a1, 1e-12f));
        orow[21 + jj] = ((jj & 1) ? a1 : a0) / n;
    }
    for (int c = tid; c < CS_; c += 256) orow[35 + c] = s[(size_t)i*CS_ + c];
}

// =======================================================================
static void gemm(const float* A, const _Float16* Wt, const float* b, const float* add,
                 float* out, int M, int K, int N, int flags, hipStream_t st)
{
    dim3 bl(128);
    if (N % 64 == 0) {
        dim3 g(N / 64, M / 64);
        if (add) {
            if (flags == 1) wmma_gemm_t<4,1,true ><<<g,bl,0,st>>>(A,Wt,b,add,out,M,K,N);
            else            wmma_gemm_t<4,0,true ><<<g,bl,0,st>>>(A,Wt,b,add,out,M,K,N);
        } else {
            if      (flags == 1) wmma_gemm_t<4,1,false><<<g,bl,0,st>>>(A,Wt,b,add,out,M,K,N);
            else if (flags == 2) wmma_gemm_t<4,2,false><<<g,bl,0,st>>>(A,Wt,b,add,out,M,K,N);
            else if (flags == 3) wmma_gemm_t<4,3,false><<<g,bl,0,st>>>(A,Wt,b,add,out,M,K,N);
            else                 wmma_gemm_t<4,0,false><<<g,bl,0,st>>>(A,Wt,b,add,out,M,K,N);
        }
    } else { // N multiple of 16 only (qp: 144, kvp: 432); flags==0, no add
        dim3 g(N / 16, M / 64);
        wmma_gemm_t<1,0,false><<<g,bl,0,st>>>(A,Wt,b,add,out,M,K,N);
    }
}

static void wconv(const float* w, _Float16* wt, int K, int N, hipStream_t st)
{
    int n = K * N;
    wconv_kernel<<<(n + 255) / 256, 256, 0, st>>>(w, wt, K, N);
}

extern "C" void kernel_launch(void* const* d_in, const int* in_sizes, int n_in,
                              void* d_out, int out_size, void* d_ws, size_t ws_size,
                              hipStream_t stream)
{
    const float* s_initial = (const float*)d_in[0];
    const float* z         = (const float*)d_in[1];
    /* d_in[2] aatype unused */
    const float* seq_mask  = (const float*)d_in[3];
    const float* ln_s_g = (const float*)d_in[4];
    const float* ln_s_b = (const float*)d_in[5];
    const float* ln_z_g = (const float*)d_in[6];
    const float* ln_z_b = (const float*)d_in[7];
    const float* lin_in_w = (const float*)d_in[8];
    const float* lin_in_b = (const float*)d_in[9];
    const float* q_w  = (const float*)d_in[10];
    const float* q_b  = (const float*)d_in[11];
    const float* kv_w = (const float*)d_in[12];
    const float* kv_b = (const float*)d_in[13];
    const float* bias_w = (const float*)d_in[14];
    const float* bias_b = (const float*)d_in[15];
    const float* qp_w  = (const float*)d_in[16];
    const float* qp_b  = (const float*)d_in[17];
    const float* kvp_w = (const float*)d_in[18];
    const float* kvp_b = (const float*)d_in[19];
    const float* gamma = (const float*)d_in[20];
    const float* out_w = (const float*)d_in[21];
    const float* out_b = (const float*)d_in[22];
    const float* ln_ipa_g = (const float*)d_in[23];
    const float* ln_ipa_b = (const float*)d_in[24];
    const float* t1_w = (const float*)d_in[25];
    const float* t1_b = (const float*)d_in[26];
    const float* t2_w = (const float*)d_in[27];
    const float* t2_b = (const float*)d_in[28];
    const float* t3_w = (const float*)d_in[29];
    const float* t3_b = (const float*)d_in[30];
    const float* tln_g = (const float*)d_in[31];
    const float* tln_b = (const float*)d_in[32];
    const float* bb_w = (const float*)d_in[33];
    const float* bb_b = (const float*)d_in[34];
    const float* ar_in_w   = (const float*)d_in[35];
    const float* ar_in_b   = (const float*)d_in[36];
    const float* ar_init_w = (const float*)d_in[37];
    const float* ar_init_b = (const float*)d_in[38];
    const float* b0_w1 = (const float*)d_in[39];
    const float* b0_b1 = (const float*)d_in[40];
    const float* b0_w2 = (const float*)d_in[41];
    const float* b0_b2 = (const float*)d_in[42];
    const float* b1_w1 = (const float*)d_in[43];
    const float* b1_b1 = (const float*)d_in[44];
    const float* b1_w2 = (const float*)d_in[45];
    const float* b1_b2 = (const float*)d_in[46];
    const float* ar_out_w = (const float*)d_in[47];
    const float* ar_out_b = (const float*)d_in[48];
    float* out = (float*)d_out;

    // ---- workspace layout (float units; f16 allocations ride the same allocator) ----
    float* wsf = (float*)d_ws;
    size_t off = 0;
    auto alloc = [&](size_t n) { float* p = wsf + off; off += (n + 63) & ~size_t(63); return p; };
    auto halloc = [&](size_t n) { return (_Float16*)alloc((n + 1) / 2); };
    float* si     = alloc((size_t)NRES * CS_);
    float* sB     = alloc((size_t)NRES * CS_);
    float* s2     = alloc((size_t)NRES * CS_);
    float* h1     = alloc((size_t)NRES * CS_);
    float* h2     = alloc((size_t)NRES * CS_);
    float* zm     = alloc((size_t)NRES * NRES);
    float* zr     = alloc((size_t)NRES * NRES);
    float* gW     = alloc(128 * 12);
    float* colGW  = alloc(64);
    float* bW     = alloc(64);
    float* biasB  = alloc((size_t)12 * NRES * NRES);
    float* aBuf   = alloc((size_t)12 * NRES * NRES);
    float* qB     = alloc((size_t)NRES * 192);
    float* kvB    = alloc((size_t)NRES * 384);
    float* qpLin  = alloc((size_t)NRES * 144);
    float* kvpLin = alloc((size_t)NRES * 432);
    float* qpPts  = alloc((size_t)NRES * 48 * 3);
    float* kvpPts = alloc((size_t)NRES * 144 * 3);
    float* sqq    = alloc((size_t)NRES * 12);
    float* sqk    = alloc((size_t)NRES * 12);
    float* oB     = alloc((size_t)NRES * 192);
    float* opB    = alloc((size_t)NRES * 288);
    float* opairB = alloc((size_t)NRES * 1536);
    float* catB   = alloc((size_t)NRES * CAT_);
    float* quat   = alloc((size_t)NRES * 4);
    float* trans  = alloc((size_t)NRES * 3);
    float* Rbuf   = alloc((size_t)NRES * 9);
    float* arInit = alloc((size_t)NRES * CRES_);
    float* aRes   = alloc((size_t)NRES * CRES_);
    float* aRes2  = alloc((size_t)NRES * CRES_);
    float* hRes   = alloc((size_t)NRES * CRES_);
    float* angB   = alloc((size_t)NRES * 14);
    // f16 transposed weights [N][K]
    _Float16* lin_in_t = halloc((size_t)CS_ * CS_);
    _Float16* q_t      = halloc((size_t)CS_ * 192);
    _Float16* kv_t     = halloc((size_t)CS_ * 384);
    _Float16* qp_t     = halloc((size_t)CS_ * 144);
    _Float16* kvp_t    = halloc((size_t)CS_ * 432);
    _Float16* out_t    = halloc((size_t)CAT_ * CS_);
    _Float16* t1_t     = halloc((size_t)CS_ * CS_);
    _Float16* t2_t     = halloc((size_t)CS_ * CS_);
    _Float16* t3_t     = halloc((size_t)CS_ * CS_);
    _Float16* ar_in_t  = halloc((size_t)CS_ * CRES_);
    _Float16* ar_ini_t = halloc((size_t)CS_ * CRES_);
    _Float16* b0w1_t   = halloc((size_t)CRES_ * CRES_);
    _Float16* b0w2_t   = halloc((size_t)CRES_ * CRES_);
    _Float16* b1w1_t   = halloc((size_t)CRES_ * CRES_);
    _Float16* b1w2_t   = halloc((size_t)CRES_ * CRES_);
    (void)ws_size; (void)n_in; (void)in_sizes; (void)out_size;

    // ---- setup (once): weight conversion + z stats + pair bias + initial projections ----
    wconv(lin_in_w, lin_in_t, CS_, CS_, stream);
    wconv(q_w,      q_t,      CS_, 192, stream);
    wconv(kv_w,     kv_t,     CS_, 384, stream);
    wconv(qp_w,     qp_t,     CS_, 144, stream);
    wconv(kvp_w,    kvp_t,    CS_, 432, stream);
    wconv(out_w,    out_t,    CAT_, CS_, stream);
    wconv(t1_w,     t1_t,     CS_, CS_, stream);
    wconv(t2_w,     t2_t,     CS_, CS_, stream);
    wconv(t3_w,     t3_t,     CS_, CS_, stream);
    wconv(ar_in_w,  ar_in_t,  CS_, CRES_, stream);
    wconv(ar_init_w,ar_ini_t, CS_, CRES_, stream);
    wconv(b0_w1,    b0w1_t,   CRES_, CRES_, stream);
    wconv(b0_w2,    b0w2_t,   CRES_, CRES_, stream);
    wconv(b1_w1,    b1w1_t,   CRES_, CRES_, stream);
    wconv(b1_w2,    b1w2_t,   CRES_, CRES_, stream);

    ln_kernel<<<NRES, 256, 0, stream>>>(s_initial, ln_s_g, ln_s_b, si, CS_);
    zstats_kernel<<<NRES * NRES, 128, 0, stream>>>(z, zm, zr);
    biasprep_kernel<<<1, 256, 0, stream>>>(bias_w, bias_b, ln_z_g, ln_z_b, gW, colGW, bW);
    bias_kernel<<<(NRES * NRES) / 8, 256, 0, stream>>>(z, zm, zr, gW, colGW, bW, biasB);
    gemm(si, lin_in_t, lin_in_b, nullptr, sB, NRES, CS_, CS_, 0, stream);
    gemm(si, ar_ini_t, ar_init_b, nullptr, arInit, NRES, CS_, CRES_, 1, stream); // relu(si)
    init_state_kernel<<<16, 32, 0, stream>>>(quat, trans);

    for (int blk = 0; blk < 8; ++blk) {
        rot_kernel<<<16, 32, 0, stream>>>(quat, Rbuf);
        // projections
        gemm(sB, q_t,   q_b,   nullptr, qB,     NRES, CS_, 192, 0, stream);
        gemm(sB, kv_t,  kv_b,  nullptr, kvB,    NRES, CS_, 384, 0, stream);
        gemm(sB, qp_t,  qp_b,  nullptr, qpLin,  NRES, CS_, 144, 0, stream);
        gemm(sB, kvp_t, kvp_b, nullptr, kvpLin, NRES, CS_, 432, 0, stream);
        pts_kernel<<<NRES, 160, 0, stream>>>(qpLin,  Rbuf, trans, qpPts,  sqq, 48,  4);
        pts_kernel<<<NRES, 160, 0, stream>>>(kvpLin, Rbuf, trans, kvpPts, sqk, 144, 12);
        // attention
        attn_kernel<<<dim3(NRES, 12), 256, 0, stream>>>(qB, kvB, qpPts, kvpPts, sqq, sqk,
                                                        biasB, gamma, seq_mask, aBuf, oB, opB);
        opair_kernel<<<NRES, 256, 0, stream>>>(z, zm, zr, aBuf, ln_z_g, ln_z_b, opairB);
        cat_kernel<<<NRES, 256, 0, stream>>>(oB, opB, opairB, Rbuf, trans, catB);
        // output projection + residual + LN
        gemm(catB, out_t, out_b, sB, s2, NRES, CAT_, CS_, 0, stream);
        ln_kernel<<<NRES, 256, 0, stream>>>(s2, ln_ipa_g, ln_ipa_b, sB, CS_);
        // transition
        gemm(sB, t1_t, t1_b, nullptr, h1, NRES, CS_, CS_, 2, stream);
        gemm(h1, t2_t, t2_b, nullptr, h2, NRES, CS_, CS_, 2, stream);
        gemm(h2, t3_t, t3_b, sB,      s2, NRES, CS_, CS_, 0, stream);
        ln_kernel<<<NRES, 256, 0, stream>>>(s2, tln_g, tln_b, sB, CS_);
        // backbone frame update (uses R from top of loop)
        bb_kernel<<<NRES, 192, 0, stream>>>(sB, bb_w, bb_b, Rbuf, quat, trans);
        // angle resnet
        gemm(sB, ar_in_t, ar_in_b, arInit, aRes, NRES, CS_, CRES_, 1, stream);
        gemm(aRes,  b0w1_t, b0_b1, nullptr, hRes,  NRES, CRES_, CRES_, 3, stream);
        gemm(hRes,  b0w2_t, b0_b2, aRes,    aRes2, NRES, CRES_, CRES_, 0, stream);
        gemm(aRes2, b1w1_t, b1_b1, nullptr, hRes,  NRES, CRES_, CRES_, 3, stream);
        gemm(hRes,  b1w2_t, b1_b2, aRes2,   aRes,  NRES, CRES_, CRES_, 0, stream);
        smallgemm_relu_kernel<<<NRES, 448, 0, stream>>>(aRes, ar_out_w, ar_out_b, angB, CRES_, 14);
        // emit
        write_out_kernel<<<NRES, 256, 0, stream>>>(quat, trans, angB, sB, out, blk);
    }
}
